// AdaSyncSSMMixer_44461501448890
// MI455X (gfx1250) — matrix operand order repaired and checked by reading, not compiled, not run on hardware
//
#include <hip/hip_runtime.h>
#include <math.h>

typedef __bf16 bf16;
typedef __attribute__((ext_vector_type(16))) __bf16 v16bf;
typedef __attribute__((ext_vector_type(8)))  __bf16 v8bf;
typedef __attribute__((ext_vector_type(8)))  float  v8f;

#define BATCH 4
#define SEQ   2048
#define HIDN  2048
#define NH    32
#define NST   64
#define HD    64
#define KCONV 4

// ---------- WMMA helpers (CDNA5 16x16x32 bf16, f32 accumulate) ----------

__device__ __forceinline__ v8f wmma_bf16(v16bf a, v16bf b, v8f c) {
  // 8 args: (neg_a, A, neg_b, B, c_mod, C, reuse_a, reuse_b)
  return __builtin_amdgcn_wmma_f32_16x16x32_bf16(false, a, false, b, (short)0, c,
                                                 false, false);
}

// A fragment: 16(M) x 32(K) bf16, row-major source with leading dim ld.
// Lane r=lane&15 holds row (row0+r); hi=lane>>4 selects K-halves:
//   elems 0..15 = K = k0 + hi*16 .. +15 (contiguous)
__device__ __forceinline__ v16bf load_a_frag(const bf16* __restrict__ base, int ld,
                                             int row0, int k0, int lane) {
  int r = lane & 15, hi = lane >> 4;
  const bf16* p = base + (size_t)(row0 + r) * ld + k0 + hi * 16;
  v8bf lo = *(const v8bf*)p;
  v8bf hv = *(const v8bf*)(p + 8);
  v16bf o;
#pragma unroll
  for (int i = 0; i < 8; ++i) { o[i] = lo[i]; o[i + 8] = hv[i]; }
  return o;
}

// B fragment for NT gemm: weight W is (Ncols x K) row-major; Bmat[k][n]=W[n][k].
// Lane c=lane&15 -> column (col0+c); hi selects K 0..15 vs 16..31 (contiguous).
__device__ __forceinline__ v16bf load_b_frag(const bf16* __restrict__ base, int ld,
                                             int col0, int k0, int lane) {
  int c = lane & 15, hi = lane >> 4;
  const bf16* p = base + (size_t)(col0 + c) * ld + k0 + hi * 16;
  return *(const v16bf*)p;
}

// ---------- fp32 -> bf16 conversion ----------

__global__ __launch_bounds__(256) void cvt_f32_bf16(const float* __restrict__ in,
                                                    bf16* __restrict__ out, int n) {
  int i = (blockIdx.x * 256 + threadIdx.x) * 8;
  if (i + 8 <= n) {
#pragma unroll
    for (int j = 0; j < 8; ++j) out[i + j] = (bf16)in[i + j];
  } else {
    for (int j = 0; j < 8 && i + j < n; ++j) out[i + j] = (bf16)in[i + j];
  }
}

// ---------- big NT GEMM: C(MxNc) = A(MxK) * B(NcxK)^T ----------
// block = 256 thr (8 waves), tile 128x128; wave -> 32(M) x 64(N).

__global__ __launch_bounds__(256) void gemm_bf16_nt(const bf16* __restrict__ A,
                                                    const bf16* __restrict__ B,
                                                    float* __restrict__ C,
                                                    int M, int Nc, int Kd) {
  int lane = threadIdx.x & 31;
  int wave = threadIdx.x >> 5;
  int wm = wave >> 1, wn = wave & 1;
  int mbase = blockIdx.y * 128 + wm * 32;
  int nbase = blockIdx.x * 128 + wn * 64;
  v8f acc[2][4];
#pragma unroll
  for (int mt = 0; mt < 2; ++mt)
#pragma unroll
    for (int nt = 0; nt < 4; ++nt)
#pragma unroll
      for (int i = 0; i < 8; ++i) acc[mt][nt][i] = 0.f;

  for (int k0 = 0; k0 < Kd; k0 += 32) {
    v16bf a0 = load_a_frag(A, Kd, mbase, k0, lane);
    v16bf a1 = load_a_frag(A, Kd, mbase + 16, k0, lane);
#pragma unroll
    for (int nt = 0; nt < 4; ++nt) {
      v16bf b = load_b_frag(B, Kd, nbase + nt * 16, k0, lane);
      acc[0][nt] = wmma_bf16(a0, b, acc[0][nt]);
      acc[1][nt] = wmma_bf16(a1, b, acc[1][nt]);
    }
  }
  int r = lane & 15, hi = lane >> 4;
#pragma unroll
  for (int mt = 0; mt < 2; ++mt)
#pragma unroll
    for (int nt = 0; nt < 4; ++nt)
#pragma unroll
      for (int v = 0; v < 8; ++v) {
        int row = mbase + mt * 16 + v + (hi ? 8 : 0);
        int col = nbase + nt * 16 + r;
        C[(size_t)row * Nc + col] = acc[mt][nt][v];
      }
}

// ---------- grouped causal conv (K=4) -> dB = (conv+bias)*dt, via WMMA ----------
// one block per (b,h); weights -> LDS as bf16 [k][n][d] (32 KB)

__global__ __launch_bounds__(256) void conv_db_kernel(const bf16* __restrict__ xbf,
                                                      const float* __restrict__ convw,
                                                      const float* __restrict__ convb,
                                                      const float* __restrict__ dt,
                                                      float* __restrict__ dB) {
  __shared__ alignas(64) bf16 lds_w[KCONV][NST][HD];
  __shared__ float lds_bias[NST];
  int b = blockIdx.x >> 5;
  int h = blockIdx.x & 31;
  int tid = threadIdx.x;
  // conv_w[(h*64+n)][d][k] -> lds_w[k][n][d]; linear idx = n*256 + d*4 + k
  for (int idx = tid; idx < NST * HD * KCONV; idx += 256) {
    int k = idx & 3;
    int d = (idx >> 2) & 63;
    int n = idx >> 8;
    lds_w[k][n][d] = (bf16)convw[(size_t)h * (NST * HD * KCONV) + idx];
  }
  if (tid < NST) lds_bias[tid] = convb[h * NST + tid];
  __syncthreads();

  int lane = tid & 31, wave = tid >> 5;
  int r = lane & 15, hi = lane >> 4;
  const bf16* xb = xbf + (size_t)b * SEQ * HIDN + h * HD;  // x[b][l][h*64 + *]

  for (int tile = wave; tile < SEQ / 16; tile += 8) {
    int l0 = tile * 16;
    v8f acc[4];
#pragma unroll
    for (int nt = 0; nt < 4; ++nt)
#pragma unroll
      for (int i = 0; i < 8; ++i) acc[nt][i] = 0.f;

#pragma unroll
    for (int k = 0; k < KCONV; ++k) {
      int row = l0 + r + k - 3;  // causal shift (pad K-1 = 3)
#pragma unroll
      for (int dbk = 0; dbk < 2; ++dbk) {
        int d0 = dbk * 32;
        v16bf a;
        if (row >= 0) {
          const bf16* p = xb + (size_t)row * HIDN + d0 + hi * 16;
          v8bf lo = *(const v8bf*)p;
          v8bf hv = *(const v8bf*)(p + 8);
#pragma unroll
          for (int i = 0; i < 8; ++i) { a[i] = lo[i]; a[i + 8] = hv[i]; }
        } else {
#pragma unroll
          for (int i = 0; i < 16; ++i) a[i] = (bf16)0.0f;
        }
#pragma unroll
        for (int nt = 0; nt < 4; ++nt) {
          const bf16* p = &lds_w[k][nt * 16 + r][d0 + hi * 16];
          v16bf bb = *(const v16bf*)p;
          acc[nt] = wmma_bf16(a, bb, acc[nt]);
        }
      }
    }
#pragma unroll
    for (int nt = 0; nt < 4; ++nt)
#pragma unroll
      for (int v = 0; v < 8; ++v) {
        int l = l0 + v + (hi ? 8 : 0);
        int n = nt * 16 + r;
        float val = (acc[nt][v] + lds_bias[n]) * dt[b * SEQ + l];
        dB[(((size_t)(b * NH + h)) * SEQ + l) * NST + n] = val;
      }
  }
}

// ---------- sequential selective scan (fp32 VALU), states out as bf16 ----------
// one block per (b,h), 64 threads (2 waves).
// Thread m keeps row A_w[h][m][:] (contiguous in global) in 64 VGPRs; the
// state vector lives in LDS, double-buffered -> one barrier per step.

__global__ __launch_bounds__(64) void scan_kernel(const float* __restrict__ A_w,
                                                  const float* __restrict__ dt,
                                                  const float* __restrict__ dB,
                                                  bf16* __restrict__ stbf) {
  __shared__ float svA[NST];
  __shared__ float svB[NST];
  int b = blockIdx.x >> 5;
  int h = blockIdx.x & 31;
  int m = threadIdx.x;

  const float* Ah = A_w + (size_t)h * NST * NST + (size_t)m * NST;
  float areg[NST];
#pragma unroll
  for (int n = 0; n < NST; ++n) areg[n] = Ah[n];

  svA[m] = 0.f;
  __syncthreads();

  const float* dBp = dB + ((size_t)(b * NH + h)) * SEQ * NST;
  bf16* stp = stbf + ((size_t)(b * NH + h)) * SEQ * NST;
  const float* dtp = dt + b * SEQ;

  float* cur = svA;
  float* nxt = svB;
  for (int l = 0; l < SEQ; ++l) {
    if (l + 1 < SEQ)
      __builtin_prefetch(dBp + (size_t)(l + 1) * NST + m, 0, 3);

    float a0 = 0.f, a1 = 0.f, a2 = 0.f, a3 = 0.f;
    const float4* svv = (const float4*)cur;
#pragma unroll
    for (int n4 = 0; n4 < NST / 4; ++n4) {
      float4 sq = svv[n4];
      a0 += sq.x * areg[4 * n4 + 0];
      a1 += sq.y * areg[4 * n4 + 1];
      a2 += sq.z * areg[4 * n4 + 2];
      a3 += sq.w * areg[4 * n4 + 3];
    }
    float alog = (a0 + a1) + (a2 + a3);
    float snew = __expf(alog * dtp[l]) * cur[m] + dBp[(size_t)l * NST + m];
    nxt[m] = snew;
    stp[(size_t)l * NST + m] = (bf16)snew;
    __syncthreads();
    float* t = cur; cur = nxt; nxt = t;
  }
}

// ---------- proj = states @ C_w^T (+ x*D) * silu(gate), via WMMA ----------

__global__ __launch_bounds__(256) void proj_gate_kernel(const bf16* __restrict__ stbf,
                                                        const float* __restrict__ C_w,
                                                        const float* __restrict__ x,
                                                        const float* __restrict__ Dv,
                                                        const float* __restrict__ gate,
                                                        float* __restrict__ hs) {
  __shared__ alignas(64) bf16 lds_c[HD][NST];  // C_w[h][d][n]
  int b = blockIdx.x >> 5;
  int h = blockIdx.x & 31;
  int tid = threadIdx.x;
  const float* Ch = C_w + (size_t)h * HD * NST;
  for (int idx = tid; idx < HD * NST; idx += 256)
    lds_c[idx >> 6][idx & 63] = (bf16)Ch[idx];
  __syncthreads();

  int lane = tid & 31, wave = tid >> 5;
  int r = lane & 15, hi = lane >> 4;
  float Dh = Dv[h];
  const bf16* stp = stbf + ((size_t)(b * NH + h)) * SEQ * NST;

  for (int tile = wave; tile < SEQ / 16; tile += 8) {
    int l0 = tile * 16;
    v8f acc[4];
#pragma unroll
    for (int dtl = 0; dtl < 4; ++dtl)
#pragma unroll
      for (int i = 0; i < 8; ++i) acc[dtl][i] = 0.f;

#pragma unroll
    for (int n0 = 0; n0 < NST; n0 += 32) {
      v16bf a = load_a_frag(stp, NST, l0, n0, lane);
#pragma unroll
      for (int dtl = 0; dtl < 4; ++dtl) {
        const bf16* p = &lds_c[dtl * 16 + r][n0 + hi * 16];
        v16bf bb = *(const v16bf*)p;
        acc[dtl] = wmma_bf16(a, bb, acc[dtl]);
      }
    }
#pragma unroll
    for (int dtl = 0; dtl < 4; ++dtl)
#pragma unroll
      for (int v = 0; v < 8; ++v) {
        int l = l0 + v + (hi ? 8 : 0);
        int col = h * HD + dtl * 16 + r;
        size_t idx = ((size_t)b * SEQ + l) * HIDN + col;
        float g = gate[idx];
        float sg = g / (1.f + __expf(-g));  // silu
        float hv = acc[dtl][v] + x[idx] * Dh;
        hs[idx] = hv * sg;
      }
  }
}

// ---------- gated RMSNorm -> bf16 ----------

__global__ __launch_bounds__(256) void rmsnorm_kernel(const float* __restrict__ hs,
                                                      const float* __restrict__ nw,
                                                      bf16* __restrict__ out) {
  __shared__ float red[256];
  size_t row = blockIdx.x;
  const float* p = hs + row * HIDN;
  bf16* o = out + row * HIDN;
  int tid = threadIdx.x;
  float vals[8];
  float ss = 0.f;
#pragma unroll
  for (int j = 0; j < 8; ++j) {
    float v = p[tid + j * 256];
    vals[j] = v;
    ss += v * v;
  }
  red[tid] = ss;
  __syncthreads();
  for (int s = 128; s > 0; s >>= 1) {
    if (tid < s) red[tid] += red[tid + s];
    __syncthreads();
  }
  float scale = rsqrtf(red[0] / (float)HIDN + 1e-6f);
#pragma unroll
  for (int j = 0; j < 8; ++j) {
    int i = tid + j * 256;
    o[i] = (bf16)(nw[i] * vals[j] * scale);
  }
}

// ---------- launch ----------

extern "C" void kernel_launch(void* const* d_in, const int* in_sizes, int n_in,
                              void* d_out, int out_size, void* d_ws, size_t ws_size,
                              hipStream_t stream) {
  (void)in_sizes; (void)n_in; (void)out_size; (void)ws_size;
  const float* x      = (const float*)d_in[0];
  const float* dt     = (const float*)d_in[1];
  const float* gate_w = (const float*)d_in[2];
  const float* A_w    = (const float*)d_in[3];
  const float* conv_w = (const float*)d_in[4];
  const float* conv_b = (const float*)d_in[5];
  const float* C_w    = (const float*)d_in[6];
  const float* Dv     = (const float*)d_in[7];
  const float* norm_w = (const float*)d_in[8];
  const float* out_w  = (const float*)d_in[9];
  float* out = (float*)d_out;

  const size_t NX = (size_t)BATCH * SEQ * HIDN;  // 16,777,216
  const size_t NW = (size_t)HIDN * HIDN;         // 4,194,304

  char* ws = (char*)d_ws;
  bf16*  xbf  = (bf16*)ws;  ws += NX * sizeof(bf16);
  bf16*  gwbf = (bf16*)ws;  ws += NW * sizeof(bf16);
  bf16*  owbf = (bf16*)ws;  ws += NW * sizeof(bf16);
  float* gateb= (float*)ws; ws += NX * sizeof(float);
  float* dB   = (float*)ws; ws += NX * sizeof(float);
  bf16*  stbf = (bf16*)ws;  ws += NX * sizeof(bf16);
  float* hs   = (float*)ws; ws += NX * sizeof(float);
  bf16*  hsn  = (bf16*)ws;  ws += NX * sizeof(bf16);

  cvt_f32_bf16<<<(int)(NX / 2048), 256, 0, stream>>>(x, xbf, (int)NX);
  cvt_f32_bf16<<<(int)(NW / 2048), 256, 0, stream>>>(gate_w, gwbf, (int)NW);
  cvt_f32_bf16<<<(int)(NW / 2048), 256, 0, stream>>>(out_w, owbf, (int)NW);

  dim3 ggrid(HIDN / 128, (BATCH * SEQ) / 128);
  gemm_bf16_nt<<<ggrid, 256, 0, stream>>>(xbf, gwbf, gateb, BATCH * SEQ, HIDN, HIDN);

  conv_db_kernel<<<BATCH * NH, 256, 0, stream>>>(xbf, conv_w, conv_b, dt, dB);
  scan_kernel<<<BATCH * NH, 64, 0, stream>>>(A_w, dt, dB, stbf);
  proj_gate_kernel<<<BATCH * NH, 256, 0, stream>>>(stbf, C_w, x, Dv, gateb, hs);
  rmsnorm_kernel<<<BATCH * SEQ, 256, 0, stream>>>(hs, norm_w, hsn);

  gemm_bf16_nt<<<ggrid, 256, 0, stream>>>(hsn, owbf, out, BATCH * SEQ, HIDN, HIDN);
}
